// WaveletBlock_43181601194294
// MI455X (gfx1250) — matrix operand order repaired and checked by reading, not compiled, not run on hardware
//
#include <hip/hip_runtime.h>
#include <hip/hip_bf16.h>
#include <math.h>

// ---------------------------------------------------------------------------
// Types for CDNA5 WMMA (wave32)
// ---------------------------------------------------------------------------
typedef __attribute__((ext_vector_type(16))) __bf16 v16bf;
typedef __attribute__((ext_vector_type(8)))  __bf16 v8bf;
typedef __attribute__((ext_vector_type(8)))  float  v8f;

#define SEQ     4096
#define HID     256
#define BATCH_N 4096
// DWT lengths: L_out(L) = floor((L+13-8)/2)+1
#define L1C 2051
#define L2C 1029
#define TC  (2*L2C + L1C)   // 4109

// ---------------------------------------------------------------------------
// WMMA fragment helpers (layouts per CDNA5 ISA 7.12.2)
// ---------------------------------------------------------------------------
// A: 16x32 bf16 tile, row-major source with leading dim lda.
// lane<16: M=lane,   K in {0..7} U {16..23}
// lane>=16: M=lane-16, K in {8..15} U {24..31}
__device__ __forceinline__ v16bf load_a_frag(const __bf16* __restrict__ base,
                                             int lda, int lane, int k0) {
  const int m    = lane & 15;
  const int koff = (lane & 16) ? 8 : 0;
  const __bf16* p = base + m * lda + k0 + koff;
  v8bf lo = *(const v8bf*)(p);        // K = koff+0 .. koff+7
  v8bf hi = *(const v8bf*)(p + 16);   // K = koff+16 .. koff+23
  return __builtin_shufflevector(lo, hi, 0,1,2,3,4,5,6,7,8,9,10,11,12,13,14,15);
}

// B: 32x16 bf16 tile taken from a TRANSPOSED (N-major) weight matrix BT[n*Ktot+k].
// lane<16: N=n0+lane,   K in k0+{0..7} U {16..23}
// lane>=16: N=n0+lane-16, K in k0+{8..15} U {24..31}
__device__ __forceinline__ v16bf load_b_fragT(const __bf16* __restrict__ BT,
                                              int Ktot, int lane, int k0, int n0) {
  const int n  = n0 + (lane & 15);
  const int kb = k0 + ((lane & 16) ? 8 : 0);
  const __bf16* p = BT + (size_t)n * Ktot + kb;
  v8bf lo = *(const v8bf*)(p);
  v8bf hi = *(const v8bf*)(p + 16);
  return __builtin_shufflevector(lo, hi, 0,1,2,3,4,5,6,7,8,9,10,11,12,13,14,15);
}

__device__ __forceinline__ v8f wmma_bf16(v16bf a, v16bf b, v8f c) {
  // (neg_a, A, neg_b, B, c_mod, C, reuse_a, reuse_b)
  return __builtin_amdgcn_wmma_f32_16x16x32_bf16(false, a, false, b, (short)0, c,
                                                 false, false);
}

// ---------------------------------------------------------------------------
// Kernel 0: f32 -> bf16 weight conversion with transpose (src is KxN, dst NxK)
// ---------------------------------------------------------------------------
__global__ void cvtT_kernel(const float* __restrict__ src, __bf16* __restrict__ dst,
                            int K, int N) {
  int i = blockIdx.x * blockDim.x + threadIdx.x;
  if (i < K * N) {
    int k = i / N;
    int n = i - k * N;
    dst[(size_t)n * K + k] = (__bf16)src[i];
  }
}

// ---------------------------------------------------------------------------
// Kernel 1: per-row 2-level db4 DWT + adaptive pooling -> pooled (BATCH x 256)
// y[n] = sum_{k=0..7} filt[k] * x[2n+1-k]  (zero padded)
// coeffs = [cA2(1029) | cD2(1029) | cD1(2051)], pool window i: [i*T/256, ceil((i+1)*T/256))
// ---------------------------------------------------------------------------
__global__ void __launch_bounds__(256)
dwt_pool_kernel(const float* __restrict__ x,
                const float* __restrict__ dec_lo, const float* __restrict__ dec_hi,
                float* __restrict__ pooled) {
  __shared__ float sx[SEQ];
  __shared__ float sA1[L1C];
  __shared__ float sD1[L1C];
  __shared__ float sA2[L2C];
  __shared__ float sD2[L2C];

  const int row = blockIdx.x;
  const int tid = threadIdx.x;

  float lo[8], hi[8];
#pragma unroll
  for (int k = 0; k < 8; ++k) { lo[k] = dec_lo[k]; hi[k] = dec_hi[k]; }

  const float* xr = x + (size_t)row * SEQ;
  for (int i = tid; i < SEQ; i += 256) sx[i] = xr[i];
  __syncthreads();

  // level 1
  for (int n = tid; n < L1C; n += 256) {
    float a = 0.f, d = 0.f;
    const int base = 2 * n + 1;
#pragma unroll
    for (int k = 0; k < 8; ++k) {
      const int idx = base - k;
      const float v = (idx >= 0 && idx < SEQ) ? sx[idx] : 0.f;
      a += lo[k] * v;
      d += hi[k] * v;
    }
    sA1[n] = a; sD1[n] = d;
  }
  __syncthreads();

  // level 2 (on cA1)
  for (int n = tid; n < L2C; n += 256) {
    float a = 0.f, d = 0.f;
    const int base = 2 * n + 1;
#pragma unroll
    for (int k = 0; k < 8; ++k) {
      const int idx = base - k;
      const float v = (idx >= 0 && idx < L1C) ? sA1[idx] : 0.f;
      a += lo[k] * v;
      d += hi[k] * v;
    }
    sA2[n] = a; sD2[n] = d;
  }
  __syncthreads();

  // adaptive average pool: one output per thread
  {
    const int i = tid;
    const int s = (i * TC) / HID;
    const int e = ((i + 1) * TC + HID - 1) / HID;
    float sum = 0.f;
    for (int t = s; t < e; ++t) {
      const float c = (t < L2C) ? sA2[t]
                    : (t < 2 * L2C) ? sD2[t - L2C]
                    : sD1[t - 2 * L2C];
      sum += c;
    }
    pooled[(size_t)row * HID + i] = sum / (float)(e - s);
  }
}

// ---------------------------------------------------------------------------
// Kernel 2: MLP gate per 16-row tile (4 waves / block, WMMA bf16 for both GEMMs)
// h = pooled@W1+b1 ; LN ; gelu ; att = sigmoid(.@W2+b2) ; g = pooled*att (bf16)
// A-fragments are hoisted: loaded once per wave, reused for all 4 N-tiles.
// ---------------------------------------------------------------------------
__global__ void __launch_bounds__(128)
mlp_kernel(const float* __restrict__ pooled,
           const __bf16* __restrict__ W1T, const float* __restrict__ b1,
           const float* __restrict__ ln_g, const float* __restrict__ ln_b,
           const __bf16* __restrict__ W2T, const float* __restrict__ b2,
           __bf16* __restrict__ gbf) {
  __shared__ float  sp[16 * HID];   // pooled tile f32
  __shared__ __bf16 sa[16 * HID];   // pooled tile bf16 (WMMA A)
  __shared__ float  sh[16 * HID];   // h tile f32
  __shared__ __bf16 sg[16 * HID];   // gelu(LN(h)) bf16 (WMMA A)

  const int tid  = threadIdx.x;
  const int lane = tid & 31;
  const int wave = tid >> 5;
  const int rb   = blockIdx.x * 16;

  for (int i = tid; i < 16 * HID; i += 128) {
    const float v = pooled[(size_t)rb * HID + i];
    sp[i] = v;
    sa[i] = (__bf16)v;
  }
  __syncthreads();

  // ---- h = A @ W1 + b1 ------------------------------------------------------
  {
    v16bf afr[8];
#pragma unroll
    for (int kt = 0; kt < 8; ++kt) afr[kt] = load_a_frag(sa, HID, lane, kt * 32);

#pragma unroll
    for (int t = 0; t < 4; ++t) {
      const int n0 = (wave * 4 + t) * 16;
      v8f acc = {};
#pragma unroll
      for (int kt = 0; kt < 8; ++kt) {
        const v16bf b = load_b_fragT(W1T, HID, lane, kt * 32, n0);
        acc = wmma_bf16(afr[kt], b, acc);
      }
      const int n     = n0 + (lane & 15);
      const int mbase = (lane & 16) ? 8 : 0;
      const float bn  = b1[n];
#pragma unroll
      for (int i = 0; i < 8; ++i) sh[(mbase + i) * HID + n] = acc[i] + bn;
    }
  }
  __syncthreads();

  // ---- LayerNorm + exact GELU (one thread per row) --------------------------
  if (tid < 16) {
    const int r = tid;
    float s = 0.f, s2 = 0.f;
    for (int j = 0; j < HID; ++j) {
      const float v = sh[r * HID + j];
      s += v; s2 += v * v;
    }
    const float mu   = s * (1.f / HID);
    const float var  = s2 * (1.f / HID) - mu * mu;
    const float rinv = rsqrtf(var + 1e-5f);
    for (int j = 0; j < HID; ++j) {
      const float hn = (sh[r * HID + j] - mu) * rinv * ln_g[j] + ln_b[j];
      const float ge = 0.5f * hn * (1.0f + erff(hn * 0.7071067811865476f));
      sg[r * HID + j] = (__bf16)ge;
    }
  }
  __syncthreads();

  // ---- att = sigmoid(gelu @ W2 + b2); g = pooled * att ----------------------
  {
    v16bf afr[8];
#pragma unroll
    for (int kt = 0; kt < 8; ++kt) afr[kt] = load_a_frag(sg, HID, lane, kt * 32);

#pragma unroll
    for (int t = 0; t < 4; ++t) {
      const int n0 = (wave * 4 + t) * 16;
      v8f acc = {};
#pragma unroll
      for (int kt = 0; kt < 8; ++kt) {
        const v16bf b = load_b_fragT(W2T, HID, lane, kt * 32, n0);
        acc = wmma_bf16(afr[kt], b, acc);
      }
      const int n     = n0 + (lane & 15);
      const int mbase = (lane & 16) ? 8 : 0;
      const float bn  = b2[n];
#pragma unroll
      for (int i = 0; i < 8; ++i) {
        const int   m   = mbase + i;
        const float att = 1.0f / (1.0f + __expf(-(acc[i] + bn)));
        const float g   = sp[m * HID + n] * att;
        gbf[(size_t)(rb + m) * HID + n] = (__bf16)g;
      }
    }
  }
}

// ---------------------------------------------------------------------------
// Kernel 3: out = x + 0.1*(G @ Wp + bp).  G: 4096x256 bf16, WpT: 4096x256 bf16.
// Block = 8 waves, computes 16 rows x 512 cols; 8 WMMAs (K=256) per 16x16 tile.
// G tile (8 KB) staged into LDS with async global->LDS b128 copies (ASYNCcnt),
// A-fragments hoisted into registers and reused across the 4 column tiles.
// ---------------------------------------------------------------------------
__global__ void __launch_bounds__(256)
out_kernel(const float* __restrict__ x, const __bf16* __restrict__ gbf,
           const __bf16* __restrict__ WpT, const float* __restrict__ bp,
           float* __restrict__ out) {
  __shared__ __bf16 sa[16 * HID];   // G tile (WMMA A); sole LDS object -> offset 0

  const int tid  = threadIdx.x;
  const int lane = tid & 31;
  const int wave = tid >> 5;
  const int rb   = blockIdx.y * 16;
  const int cb   = blockIdx.x * 512;

  // Async-stage 16x256 bf16 (8192 B) tile: 32 B per thread (2 x b128).
  {
    const __bf16* gsrc = gbf + (size_t)rb * HID;   // uniform 64-bit base (SGPR)
    unsigned loff = (unsigned)tid * 32u;           // LDS byte addr (sa starts at 0)
    unsigned goff = (unsigned)tid * 32u;           // byte offset from gsrc
    asm volatile("global_load_async_to_lds_b128 %0, %1, %2\n\t"
                 "global_load_async_to_lds_b128 %0, %1, %2 offset:16"
                 :: "v"(loff), "v"(goff), "s"(gsrc)
                 : "memory");
    asm volatile("s_wait_asynccnt 0" ::: "memory");
  }
  __syncthreads();

  // Hoist the 8 A-fragments (K = 0..255) once per wave.
  v16bf afr[8];
#pragma unroll
  for (int kt = 0; kt < 8; ++kt) afr[kt] = load_a_frag(sa, HID, lane, kt * 32);

#pragma unroll
  for (int t = 0; t < 4; ++t) {
    const int n0 = cb + wave * 64 + t * 16;
    v8f acc = {};
#pragma unroll
    for (int kt = 0; kt < 8; ++kt) {
      const v16bf b = load_b_fragT(WpT, HID, lane, kt * 32, n0);
      acc = wmma_bf16(afr[kt], b, acc);
    }
    const int   n     = n0 + (lane & 15);
    const int   mbase = (lane & 16) ? 8 : 0;
    const float bpv   = bp[n];
#pragma unroll
    for (int i = 0; i < 8; ++i) {
      const int    row = rb + mbase + i;
      const size_t idx = (size_t)row * SEQ + n;
      out[idx] = x[idx] + 0.1f * (acc[i] + bpv);
    }
  }
}

// ---------------------------------------------------------------------------
// Host-side launcher
// ---------------------------------------------------------------------------
extern "C" void kernel_launch(void* const* d_in, const int* in_sizes, int n_in,
                              void* d_out, int out_size, void* d_ws, size_t ws_size,
                              hipStream_t stream) {
  const float* x      = (const float*)d_in[0];
  const float* dec_lo = (const float*)d_in[1];
  const float* dec_hi = (const float*)d_in[2];
  const float* W1     = (const float*)d_in[3];
  const float* b1     = (const float*)d_in[4];
  const float* ln_g   = (const float*)d_in[5];
  const float* ln_b   = (const float*)d_in[6];
  const float* W2     = (const float*)d_in[7];
  const float* b2     = (const float*)d_in[8];
  const float* Wp     = (const float*)d_in[9];
  const float* bp     = (const float*)d_in[10];
  float*       out    = (float*)d_out;

  // Workspace layout (~8.3 MB total)
  char*   ws     = (char*)d_ws;
  size_t  off    = 0;
  float*  pooled = (float*)(ws + off);  off += (size_t)BATCH_N * HID * sizeof(float);
  __bf16* gbf    = (__bf16*)(ws + off); off += (size_t)BATCH_N * HID * sizeof(__bf16);
  __bf16* W1T    = (__bf16*)(ws + off); off += (size_t)HID * HID * sizeof(__bf16);
  __bf16* W2T    = (__bf16*)(ws + off); off += (size_t)HID * HID * sizeof(__bf16);
  __bf16* WpT    = (__bf16*)(ws + off); off += (size_t)HID * SEQ * sizeof(__bf16);

  cvtT_kernel<<<(HID * HID + 255) / 256, 256, 0, stream>>>(W1, W1T, HID, HID);
  cvtT_kernel<<<(HID * HID + 255) / 256, 256, 0, stream>>>(W2, W2T, HID, HID);
  cvtT_kernel<<<(HID * SEQ + 255) / 256, 256, 0, stream>>>(Wp, WpT, HID, SEQ);

  dwt_pool_kernel<<<BATCH_N, 256, 0, stream>>>(x, dec_lo, dec_hi, pooled);

  mlp_kernel<<<BATCH_N / 16, 128, 0, stream>>>(pooled, W1T, b1, ln_g, ln_b,
                                               W2T, b2, gbf);

  out_kernel<<<dim3(SEQ / 512, BATCH_N / 16), 256, 0, stream>>>(x, gbf, WpT, bp, out);
}
